// FNN_noise_cg_26405458935822
// MI455X (gfx1250) — compile-verified, matrix operand order checked by reading, not run on hardware
//
#include <hip/hip_runtime.h>
#include <math.h>

#define NXD 32
#define NYD 32
#define NZD 32
#define NBD 16
#define LMEMD 64
#define MIDD 128
#define NSITES (NXD*NYD*NZD*NBD)   /* 524288 */
#define NTILES (NSITES/16)         /* 32768  */

typedef _Float16 v16h __attribute__((ext_vector_type(16)));
typedef _Float16 v8h  __attribute__((ext_vector_type(8)));
typedef float    v8f  __attribute__((ext_vector_type(8)));

union Frag16 { v16h v; v8h h8[2]; _Float16 h[16]; };
union AccU   { v8f v; float f[8]; };

// Hardware tanh on gfx1250 (v_tanh_f32) -- avoids the branchy OCML tanhf.
__device__ __forceinline__ float hw_tanh(float x) {
#if __has_builtin(__builtin_amdgcn_tanhf)
  return __builtin_amdgcn_tanhf(x);
#elif __has_builtin(__builtin_amdgcn_tanh_f32)
  return __builtin_amdgcn_tanh_f32(x);
#else
  return tanhf(x);
#endif
}

__device__ __forceinline__ v8f wmma16(v16h a, v16h b, v8f c) {
  // D = A(16x32 f16) x B(32x16 f16) + C(16x16 f32)
  return __builtin_amdgcn_wmma_f32_16x16x32_f16(false, a, false, b, (short)0, c,
                                                false, false);
}

// Load a 16-half B (or A) fragment from an f32 weight row.
// K pattern per documented 16-bit layout: elements 0..7 -> k0..k0+7,
// elements 8..15 -> k0+16..k0+23 (k0 = 32*chunk + 8*laneHi).
__device__ __forceinline__ v16h load_w_frag(const float* __restrict__ row, int k0) {
  float g[16];
  *(float4*)(g + 0)  = *(const float4*)(row + k0);
  *(float4*)(g + 4)  = *(const float4*)(row + k0 + 4);
  *(float4*)(g + 8)  = *(const float4*)(row + k0 + 16);
  *(float4*)(g + 12) = *(const float4*)(row + k0 + 20);
  Frag16 f;
#pragma unroll
  for (int e = 0; e < 16; ++e) f.h[e] = (_Float16)g[e];
  return f.v;
}

__global__ __launch_bounds__(256)
void fused_mlp_kernel(const float* __restrict__ x_in,
                      const float* __restrict__ weight,
                      const float* __restrict__ ag,
                      const float* __restrict__ W1,  const float* __restrict__ b1,
                      const float* __restrict__ W2,  const float* __restrict__ b2,
                      const float* __restrict__ W3,  const float* __restrict__ b3,
                      const float* __restrict__ W1n, const float* __restrict__ b1n,
                      const float* __restrict__ W2n, const float* __restrict__ b2n,
                      const float* __restrict__ W3n, const float* __restrict__ b3n,
                      const float* __restrict__ max_os_l,
                      float* __restrict__ base_out,
                      float* __restrict__ nn_out)
{
  // H1 staging: 16 sites x 256 mids (f16), rows padded to 264 halves (528 B)
  // so row starts hit distinct LDS bank offsets (528/4 = 132 dwords; 132%64=4).
  __shared__ __align__(16) _Float16 H1h[16][264];
  __shared__ float dotbuf[32];   // [mlp][site]: 2 x 16

  const int tid  = threadIdx.x;
  const int lane = tid & 31;
  const int wv   = tid >> 5;     // wave 0..7
  const int l16  = lane & 15;
  const int hi   = lane >> 4;

  const float max_os = expf(max_os_l[0]);
  const float bias3  = b3[0];
  const float bias3n = b3n[0];

  // ---- Persistent weight fragments (registers, loaded once per block) ----
  // Layer 1: concatenated 256 outputs; wave w owns column tiles 2w, 2w+1.
  v16h b1f[2][2];
  float bias1[2];
#pragma unroll
  for (int j = 0; j < 2; ++j) {
    const int tcat = 2 * wv + j;                 // 0..15
    const float* Wsel = (tcat < 8) ? W1 : W1n;
    const float* bsel = (tcat < 8) ? b1 : b1n;
    const int row = 16 * (tcat & 7) + l16;       // 0..127
    bias1[j] = bsel[row];
#pragma unroll
    for (int c = 0; c < 2; ++c)
      b1f[j][c] = load_w_frag(Wsel + (size_t)row * LMEMD, 32 * c + 8 * hi);
  }
  // Layer 2: waves 0-3 -> MLP 0, waves 4-7 -> MLP 1; 2 column tiles each.
  const int p = wv >> 2;
  const float* W2sel = p ? W2n : W2;
  const float* b2sel = p ? b2n : b2;
  const float* W3sel = p ? W3n : W3;
  v16h b2f[2][4];
  float bias2[2], w3l[2];
#pragma unroll
  for (int j = 0; j < 2; ++j) {
    const int tl = (2 * wv + j) & 7;             // local tile 0..7
    const int row = 16 * tl + l16;               // 0..127
    bias2[j] = b2sel[row];
    w3l[j]   = W3sel[row];
#pragma unroll
    for (int c = 0; c < 4; ++c)
      b2f[j][c] = load_w_frag(W2sel + (size_t)row * MIDD, 32 * c + 8 * hi);
  }

  // Loop-invariant per-lane slices of weight / ag vectors (compiler keeps in
  // registers; k pattern matches the A-fragment element order).
  float wl0[16], ql0[16], wl1[16], ql1[16];
  {
    const int k0 = 8 * hi;
    *(float4*)(wl0 + 0)  = *(const float4*)(weight + k0);
    *(float4*)(wl0 + 4)  = *(const float4*)(weight + k0 + 4);
    *(float4*)(wl0 + 8)  = *(const float4*)(weight + k0 + 16);
    *(float4*)(wl0 + 12) = *(const float4*)(weight + k0 + 20);
    *(float4*)(ql0 + 0)  = *(const float4*)(ag + k0);
    *(float4*)(ql0 + 4)  = *(const float4*)(ag + k0 + 4);
    *(float4*)(ql0 + 8)  = *(const float4*)(ag + k0 + 16);
    *(float4*)(ql0 + 12) = *(const float4*)(ag + k0 + 20);
    *(float4*)(wl1 + 0)  = *(const float4*)(weight + 32 + k0);
    *(float4*)(wl1 + 4)  = *(const float4*)(weight + 32 + k0 + 4);
    *(float4*)(wl1 + 8)  = *(const float4*)(weight + 32 + k0 + 16);
    *(float4*)(wl1 + 12) = *(const float4*)(weight + 32 + k0 + 20);
    *(float4*)(ql1 + 0)  = *(const float4*)(ag + 32 + k0);
    *(float4*)(ql1 + 4)  = *(const float4*)(ag + 32 + k0 + 4);
    *(float4*)(ql1 + 8)  = *(const float4*)(ag + 32 + k0 + 16);
    *(float4*)(ql1 + 12) = *(const float4*)(ag + 32 + k0 + 20);
  }

  if (tid < 32) dotbuf[tid] = 0.0f;

  for (int tile = blockIdx.x; tile < NTILES; tile += gridDim.x) {
    const int site = tile * 16 + l16;            // lane's site (m = l16)
    const float* xr = x_in + (size_t)site * LMEMD;

    // ---- A fragments (x_w = x_in * weight) + AR partial dot ----
    float ar = 0.0f;
    v16h a1[2];
#pragma unroll
    for (int c = 0; c < 2; ++c) {
      const int k0 = 32 * c + 8 * hi;
      const float* wl = c ? wl1 : wl0;
      const float* ql = c ? ql1 : ql0;
      float g[16];
      *(float4*)(g + 0)  = *(const float4*)(xr + k0);
      *(float4*)(g + 4)  = *(const float4*)(xr + k0 + 4);
      *(float4*)(g + 8)  = *(const float4*)(xr + k0 + 16);
      *(float4*)(g + 12) = *(const float4*)(xr + k0 + 20);
      Frag16 f;
#pragma unroll
      for (int e = 0; e < 16; ++e) {
        f.h[e] = (_Float16)(g[e] * wl[e]);
        ar += g[e] * ql[e];
      }
      a1[c] = f.v;
    }
    ar += __shfl_xor(ar, 16, 32);                // lanes L, L+16 share site m

    // ---- Layer 1: H1 = tanh(Xw * W1cat^T + b1cat), store f16 to LDS ----
#pragma unroll
    for (int j = 0; j < 2; ++j) {
      AccU acc;
#pragma unroll
      for (int v = 0; v < 8; ++v) acc.f[v] = bias1[j];
      acc.v = wmma16(a1[0], b1f[j][0], acc.v);
      acc.v = wmma16(a1[1], b1f[j][1], acc.v);
      const int tcat = 2 * wv + j;
#pragma unroll
      for (int v = 0; v < 8; ++v) {
        const float hval = hw_tanh(acc.f[v]);
        H1h[v + 8 * hi][16 * tcat + l16] = (_Float16)hval;   // [site][mid]
      }
    }
    __syncthreads();

    // ---- Layer 2: A from LDS (this wave's MLP half) ----
    v16h a2[4];
    {
      const _Float16* hrow = &H1h[l16][p * 128];
#pragma unroll
      for (int c = 0; c < 4; ++c) {
        const int k0 = 32 * c + 8 * hi;
        Frag16 f;
        f.h8[0] = *(const v8h*)(hrow + k0);
        f.h8[1] = *(const v8h*)(hrow + k0 + 16);
        a2[c] = f.v;
      }
    }
    float part[8];
#pragma unroll
    for (int v = 0; v < 8; ++v) part[v] = 0.0f;
#pragma unroll
    for (int j = 0; j < 2; ++j) {
      AccU acc;
#pragma unroll
      for (int v = 0; v < 8; ++v) acc.f[v] = bias2[j];
#pragma unroll
      for (int c = 0; c < 4; ++c) acc.v = wmma16(a2[c], b2f[j][c], acc.v);
#pragma unroll
      for (int v = 0; v < 8; ++v) part[v] += hw_tanh(acc.f[v]) * w3l[j];
    }
#pragma unroll
    for (int v = 0; v < 8; ++v)
      atomicAdd(&dotbuf[p * 16 + 8 * hi + v], part[v]);      // m = v + 8*hi
    __syncthreads();

    // ---- Layer 3 finish + AR: wave 0 writes staging buffers ----
    if (wv == 0) {
      if (lane < 16) {
        const float d0 = dotbuf[lane];
        const float d1 = dotbuf[16 + lane];
        base_out[site] = hw_tanh(d0 + bias3) * max_os + ar;  // site uses l16==lane
        nn_out[site]   = hw_tanh(d1 + bias3n) * max_os;
      }
      dotbuf[lane] = 0.0f;     // same-wave LDS ops stay in order: reads precede
    }
    __syncthreads();
  }
}

__global__ __launch_bounds__(256)
void stencil_kernel(const float* __restrict__ base,
                    const float* __restrict__ nnb,
                    const float* __restrict__ sigma_l,
                    float* __restrict__ out)
{
  const int idx = blockIdx.x * blockDim.x + threadIdx.x;
  if (idx >= NSITES) return;
  const int b = idx & 15;
  const int z = (idx >> 4)  & 31;
  const int y = (idx >> 9)  & 31;
  const int x = (idx >> 14) & 31;
  const int xm = (x + 31) & 31, xp = (x + 1) & 31;
  const int ym = (y + 31) & 31, yp = (y + 1) & 31;
  const int zm = (z + 31) & 31, zp = (z + 1) & 31;
#define SITE(X, Y, Z) nnb[((X) << 14) + ((Y) << 9) + ((Z) << 4) + b]
  // roll(a, s, ax)[i] = a[i - s]: shifts (+-1 on x|y) x (+-1 on z)
  const float s = SITE(xm, y, zm) + SITE(xm, y, zp)
                + SITE(xp, y, zm) + SITE(xp, y, zp)
                + SITE(x, ym, zm) + SITE(x, ym, zp)
                + SITE(x, yp, zm) + SITE(x, yp, zp);
#undef SITE
  out[idx]          = base[idx] + s * 0.125f;
  out[NSITES + idx] = expf(sigma_l[0]);
}

extern "C" void kernel_launch(void* const* d_in, const int* in_sizes, int n_in,
                              void* d_out, int out_size, void* d_ws, size_t ws_size,
                              hipStream_t stream) {
  (void)in_sizes; (void)n_in; (void)out_size; (void)ws_size;
  const float* x_in     = (const float*)d_in[0];
  const float* weight   = (const float*)d_in[1];
  const float* ag       = (const float*)d_in[2];
  const float* W1       = (const float*)d_in[3];
  const float* b1       = (const float*)d_in[4];
  const float* W2       = (const float*)d_in[5];
  const float* b2       = (const float*)d_in[6];
  const float* W3       = (const float*)d_in[7];
  const float* b3       = (const float*)d_in[8];
  const float* W1n      = (const float*)d_in[9];
  const float* b1n      = (const float*)d_in[10];
  const float* W2n      = (const float*)d_in[11];
  const float* b2n      = (const float*)d_in[12];
  const float* W3n      = (const float*)d_in[13];
  const float* b3n      = (const float*)d_in[14];
  const float* max_os_l = (const float*)d_in[15];
  const float* sigma_l  = (const float*)d_in[16];

  float* base_buf = (float*)d_ws;           // NSITES floats
  float* nn_buf   = base_buf + NSITES;      // NSITES floats

  fused_mlp_kernel<<<2048, 256, 0, stream>>>(
      x_in, weight, ag, W1, b1, W2, b2, W3, b3,
      W1n, b1n, W2n, b2n, W3n, b3n, max_os_l, base_buf, nn_buf);

  stencil_kernel<<<NSITES / 256, 256, 0, stream>>>(
      base_buf, nn_buf, sigma_l, (float*)d_out);
}
